// RBFEuclidean_5042291605837
// MI455X (gfx1250) — matrix operand order repaired and checked by reading, not compiled, not run on hardware
//
#include <hip/hip_runtime.h>

// ---------------------------------------------------------------------------
// RBF / squared-Euclidean:  out[b,u] = ||x_b||^2 - 2 x@w + ||w_u||^2
// x: [65536,256] f32, w: [256,1024] f32, out: [65536,1024] f32.
// f32 emulated via bf16 hi/lo split: cross ~= xh*wh + xh*wl + xl*wh,
// f32 WMMA accumulation (v_wmma_f32_16x16x32_bf16), fused norm epilogue.
// Pipeline: pre-split to workspace once, then async-to-LDS double buffering.
// ---------------------------------------------------------------------------

typedef __attribute__((ext_vector_type(16))) __bf16          v16bf;
typedef __attribute__((ext_vector_type(8)))  float           v8f;
typedef __attribute__((ext_vector_type(8)))  unsigned short  ushortx8;
typedef __attribute__((ext_vector_type(4)))  unsigned short  ushortx4;
typedef unsigned short u16;
typedef unsigned int   u32;

#define M_TOT 65536
#define N_TOT 1024
#define K_TOT 256

#define BM 128       // block tile M
#define BN 128       // block tile N
#define KC 32        // K chunk (one WMMA-K)
#define KCP 40       // padded LDS row (80B: 16B-aligned, bank-spread)
#define NCHUNK (K_TOT / KC)
#define NTHREADS 256

__device__ __forceinline__ u32  f2u(float f)  { return __builtin_bit_cast(u32, f); }
__device__ __forceinline__ float u2f(u32 u)   { return __builtin_bit_cast(float, u); }

// ---------------------------------------------------------------------------
// Fragment loaders (ISA 7.12.2 layouts)
// A (16x32 bf16): lanes 0-15 row M=lane, v0-3=K0..7, v4-7=K16..23;
//                 lanes 16-31 same rows, K8..15 / K24..31.
__device__ __forceinline__ v16bf ld_fragA(const u16* rowp, int lh) {
    ushortx8 a = *(const ushortx8*)(rowp + lh * 8);
    ushortx8 b = *(const ushortx8*)(rowp + 16 + lh * 8);
    return __builtin_bit_cast(v16bf,
        __builtin_shufflevector(a, b, 0,1,2,3,4,5,6,7,8,9,10,11,12,13,14,15));
}
// B (32x16 bf16): lane holds col N=(lane&15); lanes 0-15 K0..7 / K16..23,
// lanes 16-31 K8..15 / K24..31 (same K order as A).
__device__ __forceinline__ v16bf ld_fragB(const u16* colp, int lh) {
    ushortx8 a = *(const ushortx8*)(colp + lh * 8);
    ushortx8 b = *(const ushortx8*)(colp + 16 + lh * 8);
    return __builtin_bit_cast(v16bf,
        __builtin_shufflevector(a, b, 0,1,2,3,4,5,6,7,8,9,10,11,12,13,14,15));
}

// Async global->LDS 16B copy (per-lane addresses), tracked by ASYNCcnt.
__device__ __forceinline__ void async_ld16(u32 lds_off, const void* gaddr) {
    asm volatile("global_load_async_to_lds_b128 %0, %1, off"
                 :: "v"(lds_off), "v"(gaddr) : "memory");
}
__device__ __forceinline__ void wait_async_le8() {
    asm volatile("s_wait_asynccnt 0x8" ::: "memory");
}
__device__ __forceinline__ void wait_async_0() {
    asm volatile("s_wait_asynccnt 0x0" ::: "memory");
}

// ---------------------------------------------------------------------------
// Prep A: split x -> xh/xl (bf16 bits, row-major) + fused row norms.
// One wave per row; v_perm_b32 packs two bf16 per op.
// ---------------------------------------------------------------------------
__global__ __launch_bounds__(256) void prep_x(const float* __restrict__ x,
                                              u16* __restrict__ xh,
                                              u16* __restrict__ xl,
                                              float* __restrict__ xsq) {
    const int lane = threadIdx.x & 31;
    const size_t row = (size_t)blockIdx.x * 8 + (threadIdx.x >> 5);
    const float4* px = (const float4*)(x + row * K_TOT);
    uint2* ph = (uint2*)(xh + row * K_TOT);
    uint2* pl = (uint2*)(xl + row * K_TOT);
    float s = 0.0f;
    #pragma unroll
    for (int j = 0; j < 2; ++j) {
        float4 v = px[lane + j * 32];
        s += v.x * v.x + v.y * v.y + v.z * v.z + v.w * v.w;
        u32 a0 = f2u(v.x), a1 = f2u(v.y), a2 = f2u(v.z), a3 = f2u(v.w);
        // hi = truncate-to-bf16 (exact); lo = truncate-to-bf16(v - hi)
        u32 b0 = f2u(v.x - u2f(a0 & 0xFFFF0000u));
        u32 b1 = f2u(v.y - u2f(a1 & 0xFFFF0000u));
        u32 b2 = f2u(v.z - u2f(a2 & 0xFFFF0000u));
        u32 b3 = f2u(v.w - u2f(a3 & 0xFFFF0000u));
        uint2 hv, lv;
        hv.x = __builtin_amdgcn_perm(a1, a0, 0x07060302u);
        hv.y = __builtin_amdgcn_perm(a3, a2, 0x07060302u);
        lv.x = __builtin_amdgcn_perm(b1, b0, 0x07060302u);
        lv.y = __builtin_amdgcn_perm(b3, b2, 0x07060302u);
        ph[lane + j * 32] = hv;
        pl[lane + j * 32] = lv;
    }
    #pragma unroll
    for (int off = 16; off > 0; off >>= 1) s += __shfl_xor(s, off, 32);
    if (lane == 0) xsq[row] = s;
}

// ---------------------------------------------------------------------------
// Prep B: split + transpose w -> wth/wtl [1024][256] (K-contiguous per col)
// + fused column norms. Tiny (1 MB), one-time.
// ---------------------------------------------------------------------------
__global__ __launch_bounds__(256) void prep_w(const float* __restrict__ w,
                                              u16* __restrict__ wth,
                                              u16* __restrict__ wtl,
                                              float* __restrict__ wsq) {
    const int u = blockIdx.x * 256 + threadIdx.x;
    float s = 0.0f;
    for (int k = 0; k < K_TOT; ++k) {
        float v = w[(size_t)k * N_TOT + u];
        s += v * v;
        u32 uu = f2u(v);
        wth[(size_t)u * K_TOT + k] = (u16)(uu >> 16);
        u32 ul = f2u(v - u2f(uu & 0xFFFF0000u));
        wtl[(size_t)u * K_TOT + k] = (u16)(ul >> 16);
    }
    wsq[u] = s;
}

// ---------------------------------------------------------------------------
// Main GEMM: async double-buffered staging, pure WMMA inner loop.
// 256 threads = 8 waves (2 M x 4 N), wave tile 64x32, block tile 128x128.
// ---------------------------------------------------------------------------
__device__ __forceinline__ void stage_async(const u16* __restrict__ xh,
                                            const u16* __restrict__ xl,
                                            const u16* __restrict__ wth,
                                            const u16* __restrict__ wtl,
                                            int m0, int n0, int k0, int tid,
                                            u32 bXh, u32 bXl, u32 bWh, u32 bWl) {
    #pragma unroll
    for (int j = 0; j < 2; ++j) {
        int q   = tid * 2 + j;       // 0..511 16B-transfers per array
        int row = q >> 2;            // 4 transfers per 32-k row
        int sk  = (q & 3) * 8;       // k offset (shorts)
        u32 loff = (u32)(row * KCP + sk) * 2;
        size_t gxo = (size_t)(m0 + row) * K_TOT + k0 + sk;
        size_t gwo = (size_t)(n0 + row) * K_TOT + k0 + sk;
        async_ld16(bXh + loff, xh + gxo);
        async_ld16(bXl + loff, xl + gxo);
        async_ld16(bWh + loff, wth + gwo);
        async_ld16(bWl + loff, wtl + gwo);
    }
}

__global__ __launch_bounds__(NTHREADS) void rbf_main_async(
        const u16* __restrict__ xh, const u16* __restrict__ xl,
        const u16* __restrict__ wth, const u16* __restrict__ wtl,
        const float* __restrict__ xsq, const float* __restrict__ wsq,
        float* __restrict__ out) {
    __shared__ __align__(16) u16 sXh[2][BM][KCP];
    __shared__ __align__(16) u16 sXl[2][BM][KCP];
    __shared__ __align__(16) u16 sWh[2][BN][KCP];
    __shared__ __align__(16) u16 sWl[2][BN][KCP];
    __shared__ float sXsq[BM];
    __shared__ float sWsq[BN];

    const int m0   = blockIdx.y * BM;
    const int n0   = blockIdx.x * BN;
    const int tid  = threadIdx.x;
    const int lane = tid & 31;
    const int wid  = tid >> 5;
    const int wm   = (wid >> 2) * 64;
    const int wn   = (wid & 3) * 32;
    const int lr   = lane & 15;
    const int lh   = lane >> 4;

    const u32 bXh[2] = { (u32)(size_t)&sXh[0][0][0], (u32)(size_t)&sXh[1][0][0] };
    const u32 bXl[2] = { (u32)(size_t)&sXl[0][0][0], (u32)(size_t)&sXl[1][0][0] };
    const u32 bWh[2] = { (u32)(size_t)&sWh[0][0][0], (u32)(size_t)&sWh[1][0][0] };
    const u32 bWl[2] = { (u32)(size_t)&sWl[0][0][0], (u32)(size_t)&sWl[1][0][0] };

    if (tid < BM) sXsq[tid] = xsq[m0 + tid];
    if (tid < BN) sWsq[tid] = wsq[n0 + tid];

    v8f acc[4][2] = {};

    // prefetch chunk 0
    stage_async(xh, xl, wth, wtl, m0, n0, 0, tid, bXh[0], bXl[0], bWh[0], bWl[0]);

    for (int i = 0; i < NCHUNK; ++i) {
        const int cur = i & 1;
        if (i + 1 < NCHUNK) {
            const int nxt = (i + 1) & 1;
            stage_async(xh, xl, wth, wtl, m0, n0, (i + 1) * KC, tid,
                        bXh[nxt], bXl[nxt], bWh[nxt], bWl[nxt]);
            wait_async_le8();   // in-order completion => chunk i's 8 are done
        } else {
            wait_async_0();
        }
        __syncthreads();        // chunk i visible from all waves

        v16bf ah[4], al[4], bh[2], bl[2];
        #pragma unroll
        for (int mt = 0; mt < 4; ++mt) {
            int row = wm + mt * 16 + lr;
            ah[mt] = ld_fragA(&sXh[cur][row][0], lh);
            al[mt] = ld_fragA(&sXl[cur][row][0], lh);
        }
        #pragma unroll
        for (int nt = 0; nt < 2; ++nt) {
            int col = wn + nt * 16 + lr;
            bh[nt] = ld_fragB(&sWh[cur][col][0], lh);
            bl[nt] = ld_fragB(&sWl[cur][col][0], lh);
        }
        __syncthreads();        // all reads of buffer `cur` complete

        #pragma unroll
        for (int mt = 0; mt < 4; ++mt) {
            #pragma unroll
            for (int nt = 0; nt < 2; ++nt) {
                acc[mt][nt] = __builtin_amdgcn_wmma_f32_16x16x32_bf16(
                    false, ah[mt], false, bh[nt], (short)0, acc[mt][nt], false, false);
                acc[mt][nt] = __builtin_amdgcn_wmma_f32_16x16x32_bf16(
                    false, ah[mt], false, bl[nt], (short)0, acc[mt][nt], false, false);
                acc[mt][nt] = __builtin_amdgcn_wmma_f32_16x16x32_bf16(
                    false, al[mt], false, bh[nt], (short)0, acc[mt][nt], false, false);
            }
        }
    }

    // epilogue: out = xsq + wsq - 2*cross  (C/D: VGPR i -> row i+8*lh, col lr)
    #pragma unroll
    for (int mt = 0; mt < 4; ++mt) {
        #pragma unroll
        for (int nt = 0; nt < 2; ++nt) {
            int col_l = wn + nt * 16 + lr;
            float wsq_v = sWsq[col_l];
            size_t col = (size_t)(n0 + col_l);
            #pragma unroll
            for (int i = 0; i < 8; ++i) {
                int row_l = wm + mt * 16 + i + lh * 8;
                float v = sXsq[row_l] + wsq_v - 2.0f * acc[mt][nt][i];
                out[(size_t)(m0 + row_l) * N_TOT + col] = v;
            }
        }
    }
}

// ===========================================================================
// Fallback path (small workspace): inline split + synchronous staging.
// ===========================================================================
struct BfPair { u16 h, l; };
__device__ __forceinline__ BfPair split_bf16(float f) {
    BfPair r;
    u32 u = f2u(f);
    r.h = (u16)(u >> 16);
    u32 ul = f2u(f - u2f(u & 0xFFFF0000u));
    r.l = (u16)((ul + 0x7FFFu + ((ul >> 16) & 1u)) >> 16);
    return r;
}

__global__ __launch_bounds__(256) void rbf_row_norms(const float* __restrict__ x,
                                                     float* __restrict__ xsq) {
    const int lane = threadIdx.x & 31;
    const int row  = blockIdx.x * 8 + (threadIdx.x >> 5);
    const float* p = x + (size_t)row * K_TOT;
    float s = 0.0f;
    #pragma unroll
    for (int i = 0; i < K_TOT / 32; ++i) { float v = p[lane + i * 32]; s += v * v; }
    #pragma unroll
    for (int off = 16; off > 0; off >>= 1) s += __shfl_xor(s, off, 32);
    if (lane == 0) xsq[row] = s;
}
__global__ __launch_bounds__(256) void rbf_col_norms(const float* __restrict__ w,
                                                     float* __restrict__ wsq) {
    const int u = blockIdx.x * 256 + threadIdx.x;
    float s = 0.0f;
    for (int k = 0; k < K_TOT; ++k) { float v = w[(size_t)k * N_TOT + u]; s += v * v; }
    wsq[u] = s;
}

__global__ __launch_bounds__(NTHREADS) void rbf_main_inline(
        const float* __restrict__ x, const float* __restrict__ w,
        const float* __restrict__ xsq, const float* __restrict__ wsq,
        float* __restrict__ out) {
    __shared__ __align__(16) u16 sXh[BM][KCP];
    __shared__ __align__(16) u16 sXl[BM][KCP];
    __shared__ __align__(16) u16 sWh[BN][KCP];
    __shared__ __align__(16) u16 sWl[BN][KCP];
    __shared__ float sXsq[BM];
    __shared__ float sWsq[BN];

    const int m0 = blockIdx.y * BM, n0 = blockIdx.x * BN;
    const int tid = threadIdx.x, lane = tid & 31, wid = tid >> 5;
    const int wm = (wid >> 2) * 64, wn = (wid & 3) * 32;
    const int lr = lane & 15, lh = lane >> 4;

    if (tid < BM) sXsq[tid] = xsq[m0 + tid];
    if (tid < BN) sWsq[tid] = wsq[n0 + tid];

    v8f acc[4][2] = {};
    for (int k0 = 0; k0 < K_TOT; k0 += KC) {
        __syncthreads();
        #pragma unroll
        for (int it = 0; it < 4; ++it) {
            int q = tid + it * NTHREADS;
            int row = q >> 3, cg = (q & 7) * 4;
            const float4 v = *(const float4*)&x[(size_t)(m0 + row) * K_TOT + k0 + cg];
            ushortx4 h, l;
            BfPair p0 = split_bf16(v.x); h.x = p0.h; l.x = p0.l;
            BfPair p1 = split_bf16(v.y); h.y = p1.h; l.y = p1.l;
            BfPair p2 = split_bf16(v.z); h.z = p2.h; l.z = p2.l;
            BfPair p3 = split_bf16(v.w); h.w = p3.h; l.w = p3.l;
            *(ushortx4*)&sXh[row][cg] = h;
            *(ushortx4*)&sXl[row][cg] = l;
        }
        #pragma unroll
        for (int it = 0; it < 4; ++it) {
            int q = tid + it * NTHREADS;
            int kr = q >> 5, cg = (q & 31) * 4;
            const float4 v = *(const float4*)&w[(size_t)(k0 + kr) * N_TOT + n0 + cg];
            BfPair p0 = split_bf16(v.x); sWh[cg + 0][kr] = p0.h; sWl[cg + 0][kr] = p0.l;
            BfPair p1 = split_bf16(v.y); sWh[cg + 1][kr] = p1.h; sWl[cg + 1][kr] = p1.l;
            BfPair p2 = split_bf16(v.z); sWh[cg + 2][kr] = p2.h; sWl[cg + 2][kr] = p2.l;
            BfPair p3 = split_bf16(v.w); sWh[cg + 3][kr] = p3.h; sWl[cg + 3][kr] = p3.l;
        }
        __syncthreads();

        v16bf ah[4], al[4], bh[2], bl[2];
        #pragma unroll
        for (int mt = 0; mt < 4; ++mt) {
            int row = wm + mt * 16 + lr;
            ah[mt] = ld_fragA(&sXh[row][0], lh);
            al[mt] = ld_fragA(&sXl[row][0], lh);
        }
        #pragma unroll
        for (int nt = 0; nt < 2; ++nt) {
            int col = wn + nt * 16 + lr;
            bh[nt] = ld_fragB(&sWh[col][0], lh);
            bl[nt] = ld_fragB(&sWl[col][0], lh);
        }
        #pragma unroll
        for (int mt = 0; mt < 4; ++mt)
            #pragma unroll
            for (int nt = 0; nt < 2; ++nt) {
                acc[mt][nt] = __builtin_amdgcn_wmma_f32_16x16x32_bf16(
                    false, ah[mt], false, bh[nt], (short)0, acc[mt][nt], false, false);
                acc[mt][nt] = __builtin_amdgcn_wmma_f32_16x16x32_bf16(
                    false, ah[mt], false, bl[nt], (short)0, acc[mt][nt], false, false);
                acc[mt][nt] = __builtin_amdgcn_wmma_f32_16x16x32_bf16(
                    false, al[mt], false, bh[nt], (short)0, acc[mt][nt], false, false);
            }
    }
    #pragma unroll
    for (int mt = 0; mt < 4; ++mt)
        #pragma unroll
        for (int nt = 0; nt < 2; ++nt) {
            int col_l = wn + nt * 16 + lr;
            float wsq_v = sWsq[col_l];
            size_t col = (size_t)(n0 + col_l);
            #pragma unroll
            for (int i = 0; i < 8; ++i) {
                int row_l = wm + mt * 16 + i + lh * 8;
                out[(size_t)(m0 + row_l) * N_TOT + col] =
                    sXsq[row_l] + wsq_v - 2.0f * acc[mt][nt][i];
            }
        }
}

// ---------------------------------------------------------------------------
extern "C" void kernel_launch(void* const* d_in, const int* in_sizes, int n_in,
                              void* d_out, int out_size, void* d_ws, size_t ws_size,
                              hipStream_t stream) {
    (void)in_sizes; (void)n_in; (void)out_size;
    const float* x = (const float*)d_in[0];
    const float* w = (const float*)d_in[1];
    float* out = (float*)d_out;

    char* ws = (char*)d_ws;
    const size_t off_xsq = 0;
    const size_t off_wsq = off_xsq + (size_t)M_TOT * 4;
    const size_t off_wth = off_wsq + (size_t)N_TOT * 4;
    const size_t off_wtl = off_wth + (size_t)N_TOT * K_TOT * 2;
    const size_t off_xh  = off_wtl + (size_t)N_TOT * K_TOT * 2;
    const size_t off_xl  = off_xh + (size_t)M_TOT * K_TOT * 2;
    const size_t need    = off_xl + (size_t)M_TOT * K_TOT * 2;

    float* xsq = (float*)(ws + off_xsq);
    float* wsq = (float*)(ws + off_wsq);
    dim3 grid(N_TOT / BN, M_TOT / BM);

    if (ws_size >= need) {
        u16* wth = (u16*)(ws + off_wth);
        u16* wtl = (u16*)(ws + off_wtl);
        u16* xh  = (u16*)(ws + off_xh);
        u16* xl  = (u16*)(ws + off_xl);
        prep_x<<<M_TOT / 8, 256, 0, stream>>>(x, xh, xl, xsq);
        prep_w<<<N_TOT / 256, 256, 0, stream>>>(w, wth, wtl, wsq);
        rbf_main_async<<<grid, NTHREADS, 0, stream>>>(xh, xl, wth, wtl, xsq, wsq, out);
    } else {
        rbf_row_norms<<<M_TOT / 8, 256, 0, stream>>>(x, xsq);
        rbf_col_norms<<<N_TOT / 256, 256, 0, stream>>>(w, wsq);
        rbf_main_inline<<<grid, NTHREADS, 0, stream>>>(x, w, xsq, wsq, out);
    }
}
